// GCN_alignment_gram_47502338294136
// MI455X (gfx1250) — matrix operand order repaired and checked by reading, not compiled, not run on hardware
//
#include <hip/hip_runtime.h>
#include <hip/hip_bf16.h>

typedef __bf16 bf16_t;
typedef __attribute__((ext_vector_type(4)))  __bf16 v4bf;
typedef __attribute__((ext_vector_type(8)))  __bf16 v8bf;
typedef __attribute__((ext_vector_type(16))) __bf16 v16bf;
typedef __attribute__((ext_vector_type(8)))  float  v8f;

#define BM  128
#define BN  64
#define BKT 32
#define LDS_STRIDE 40   // 32 + 8 pad, 80B rows => every 16B chunk stays aligned

// ---------------------------------------------------------------------------
// CDNA5 async copy: global -> LDS, 16 bytes per lane, tracked by ASYNCcnt.
// vdst = 32-bit wave-relative LDS byte offset, vaddr = 64-bit global address.
// ---------------------------------------------------------------------------
__device__ __forceinline__ void async_load_b128(unsigned ldsAddr, const void* gptr) {
  asm volatile("global_load_async_to_lds_b128 %0, %1, off"
               :: "v"(ldsAddr), "v"(gptr) : "memory");
}
__device__ __forceinline__ void wait_asynccnt0() {
  asm volatile("s_wait_asynccnt 0x0" ::: "memory");
}
__device__ __forceinline__ unsigned lds_offset(const void* p) {
  return (unsigned)(unsigned long long)p;   // low 32 bits = LDS byte offset
}

// ---------------------------------------------------------------------------
// Fragment builders (ISA 7.12.2, wave32). Contiguous 128-bit LDS loads.
// A tile in LDS: row-major [m][k], stride 40.
//   lane m=lane&15, kb=(lane&16)?8:0 ; K runs kb..kb+7 and 16+kb..16+kb+7
__device__ __forceinline__ v16bf frag_a(const bf16_t* p, int lane) {
  int m  = lane & 15;
  int kb = (lane & 16) ? 8 : 0;
  const bf16_t* r = p + m * LDS_STRIDE + kb;
  v8bf lo = *(const v8bf*)r;
  v8bf hi = *(const v8bf*)(r + 16);
  return __builtin_shufflevector(lo, hi, 0, 1, 2, 3, 4, 5, 6, 7,
                                         8, 9, 10, 11, 12, 13, 14, 15);
}
// B tile in LDS stored TRANSPOSED: [n][k], stride 40.
//   lane n=lane&15, kb=(lane&16)?8:0 ; K runs kb..kb+7 and 16+kb..16+kb+7
__device__ __forceinline__ v16bf frag_b(const bf16_t* p, int lane) {
  int n  = lane & 15;
  int kb = (lane & 16) ? 8 : 0;
  const bf16_t* r = p + n * LDS_STRIDE + kb;
  v8bf lo = *(const v8bf*)r;
  v8bf hi = *(const v8bf*)(r + 16);
  return __builtin_shufflevector(lo, hi, 0, 1, 2, 3, 4, 5, 6, 7,
                                         8, 9, 10, 11, 12, 13, 14, 15);
}

#define WMMA_BF16(a, b, c) __builtin_amdgcn_wmma_f32_16x16x32_bf16( \
    false, (a), false, (b), (short)0, (c), false, false)

// ---------------------------------------------------------------------------
// Batched bf16 WMMA GEMM. Block 256 threads (8 wave32), tile 128x64, K-step 32.
// Each wave computes a 32x32 output (2x2 WMMA accumulators) -> 4 wmma / iter.
// Staging uses GLOBAL_LOAD_ASYNC_TO_LDS_B128 (ASYNCcnt) where layouts allow.
// ---------------------------------------------------------------------------
template <bool TRANSB, bool GATHER, bool BIAS, bool RELU, bool OUTF, bool OUTB>
__global__ void __launch_bounds__(256)
wmma_gemm(const bf16_t* __restrict__ Ab, long sAb, int lda,
          const bf16_t* __restrict__ Bb, long sBb, int ldb,
          float* __restrict__ Cf, bf16_t* __restrict__ Cb, long sCb, int ldc,
          const float* __restrict__ bias,
          const int* __restrict__ gatherIdx, int Kk) {
  __shared__ __align__(16) bf16_t As[BM][LDS_STRIDE];
  __shared__ __align__(16) bf16_t Bs[BN][LDS_STRIDE];   // [n][k]

  const int batch = blockIdx.z;
  const bf16_t* Aptr = Ab + (long)batch * sAb;
  const bf16_t* Bptr = Bb + (long)batch * sBb;
  const long cbase = (long)batch * sCb;

  const int n0 = blockIdx.x * BN;
  const int m0 = blockIdx.y * BM;
  const int t  = threadIdx.x;
  const int w = t >> 5, lane = t & 31;
  const int wr = w >> 1, wc = w & 1;          // 4x2 waves of 32x32 tiles

  // ---- hoist staging addresses (rows handled by a thread are K-invariant) --
  // A: 128x32 elems = 512 16B-chunks (8 bf16), 2 per thread, via async copy
  const bf16_t* aSrc[2];
  unsigned aLds[2];
#pragma unroll
  for (int i = 0; i < 2; ++i) {
    int q = i * 256 + t;
    int row = q >> 2, cb = (q & 3) * 8;
    long sr = GATHER ? (long)gatherIdx[m0 + row] : (long)(m0 + row);
    aSrc[i] = Aptr + sr * lda + cb;
    aLds[i] = lds_offset(&As[row][cb]);
  }
  // B staging
  const bf16_t* bSrcT = nullptr;
  unsigned bLds = 0;
  const bf16_t* bColBase = nullptr;
  bf16_t* bDstN[8];
  int bKN[8];
  if (TRANSB) {
    // B is [N,K]: 64x32 elems = 256 16B-chunks, 1 per thread, via async copy
    int n = t >> 2, cb = (t & 3) * 8;
    bSrcT = Bptr + (long)(n0 + n) * ldb + cb;
    bLds = lds_offset(&Bs[n][cb]);
  } else {
    // B is [K,N]: column reads (scalar), n coalesced across lanes. 8 elems/thread
    bColBase = Bptr + n0 + (t & 63);
#pragma unroll
    for (int i = 0; i < 8; ++i) {
      int k = i * 4 + (t >> 6);
      bKN[i] = k;
      bDstN[i] = &Bs[t & 63][k];
    }
  }

  v8f acc00 = {}, acc01 = {}, acc10 = {}, acc11 = {};

  for (int k0 = 0; k0 < Kk; k0 += BKT) {
    // ---- stage K-tile into LDS via async DMA (ASYNCcnt-tracked) ----
#pragma unroll
    for (int i = 0; i < 2; ++i)
      async_load_b128(aLds[i], aSrc[i] + k0);
    if (TRANSB) {
      async_load_b128(bLds, bSrcT + k0);
    } else {
#pragma unroll
      for (int i = 0; i < 8; ++i)
        *bDstN[i] = bColBase[(long)(k0 + bKN[i]) * ldb];
    }
    // prefetch next K-tile into cache while this one computes
#pragma unroll
    for (int i = 0; i < 2; ++i)
      __builtin_prefetch(aSrc[i] + k0 + BKT, 0, 1);
    if (TRANSB)
      __builtin_prefetch(bSrcT + k0 + BKT, 0, 1);

    wait_asynccnt0();
    __syncthreads();

    v16bf a0 = frag_a(&As[wr * 32][0],      lane);
    v16bf a1 = frag_a(&As[wr * 32 + 16][0], lane);
    v16bf b0 = frag_b(&Bs[wc * 32][0],      lane);
    v16bf b1 = frag_b(&Bs[wc * 32 + 16][0], lane);
    acc00 = WMMA_BF16(a0, b0, acc00);
    acc01 = WMMA_BF16(a0, b1, acc01);
    acc10 = WMMA_BF16(a1, b0, acc10);
    acc11 = WMMA_BF16(a1, b1, acc11);

    __syncthreads();
  }

  // ---- epilogue: C layout lanes 0-15 -> N=lane, M=vgpr r; lanes 16-31 M=r+8
  const int n  = lane & 15;
  const int mb = (lane & 16) ? 8 : 0;
  const v8f accs[2][2] = {{acc00, acc01}, {acc10, acc11}};
#pragma unroll
  for (int mi = 0; mi < 2; ++mi) {
#pragma unroll
    for (int ni = 0; ni < 2; ++ni) {
      const int gcol = n0 + wc * 32 + ni * 16 + n;
      float bv = 0.f;
      if constexpr (BIAS) bv = bias[gcol];
      const long rowbase = cbase + (long)(m0 + wr * 32 + mi * 16 + mb) * ldc + gcol;
      const v8f a = accs[mi][ni];
#pragma unroll
      for (int r = 0; r < 8; ++r) {
        float v = a[r];
        if constexpr (BIAS) v += bv;
        if constexpr (RELU) v = fmaxf(v, 0.f);
        if constexpr (OUTF) Cf[rowbase + (long)r * ldc] = v;
        if constexpr (OUTB) Cb[rowbase + (long)r * ldc] = (bf16_t)v;
      }
    }
  }
}

// ---------------------------------------------------------------------------
// f32 -> bf16 conversion, 4-wide (n always multiple of 4 here)
// ---------------------------------------------------------------------------
__global__ void cvt_f32_bf16(const float4* __restrict__ in, v4bf* __restrict__ out, long n4) {
  long i = (long)blockIdx.x * blockDim.x + threadIdx.x;
  long stride = (long)gridDim.x * blockDim.x;
  for (; i < n4; i += stride) {
    float4 f = in[i];
    v4bf o;
    o[0] = (bf16_t)f.x; o[1] = (bf16_t)f.y; o[2] = (bf16_t)f.z; o[3] = (bf16_t)f.w;
    out[i] = o;
  }
}

// ---------------------------------------------------------------------------
// Row/column softmax over a 128-wide axis of s[b] -> bf16 probabilities
// ---------------------------------------------------------------------------
__global__ void softmax128(const float* __restrict__ s, bf16_t* __restrict__ p,
                           long in_rs, long in_cs, long in_bs, long out_bs, int cols) {
  const int b = blockIdx.y, r = blockIdx.x, t = threadIdx.x;
  const float* src = s + (long)b * in_bs;
  float v = src[(long)r * in_rs + (long)t * in_cs];
  __shared__ float red[128];
  red[t] = v; __syncthreads();
  for (int o = 64; o > 0; o >>= 1) { if (t < o) red[t] = fmaxf(red[t], red[t + o]); __syncthreads(); }
  float mx = red[0]; __syncthreads();
  float e = __expf(v - mx);
  red[t] = e; __syncthreads();
  for (int o = 64; o > 0; o >>= 1) { if (t < o) red[t] += red[t + o]; __syncthreads(); }
  float sum = red[0];
  p[(long)b * out_bs + (long)r * cols + t] = (bf16_t)(e / sum);
}

// ---------------------------------------------------------------------------
// t = concat((X-E)^2, X*E)  -> bf16 [B*L, 512]
// ---------------------------------------------------------------------------
__global__ void build_t(const float* __restrict__ X, const float* __restrict__ E,
                        bf16_t* __restrict__ T, long n /* B*L*H */) {
  long i = (long)blockIdx.x * blockDim.x + threadIdx.x;
  if (i >= n) return;
  long row = i >> 8;            // H = 256
  int  h   = (int)(i & 255);
  float x = X[i], e = E[i];
  float d = x - e;
  T[row * 512 + h]       = (bf16_t)(d * d);
  T[row * 512 + 256 + h] = (bf16_t)(x * e);
}

// ---------------------------------------------------------------------------
// max-pool over L of relu(V + bias)   (fs=1 conv branch)
// ---------------------------------------------------------------------------
__global__ void pool_e1(const float* __restrict__ V, const float* __restrict__ bias,
                        float* __restrict__ re, int L, int K, int cbase, int reW) {
  int b = blockIdx.x, k = threadIdx.x;
  const float* vp = V + (long)b * L * K + k;
  float m = -1e30f;
  for (int l = 0; l < L; ++l) m = fmaxf(m, fmaxf(vp[(long)l * K] + bias[k], 0.f));
  re[(long)b * reW + cbase + k] = fmaxf(m, 0.f);
}

// max-pool over L-1 of relu(U0[l] + U1[l+1] + bias)   (fs=2 conv branch)
__global__ void pool_e2(const float* __restrict__ U0, const float* __restrict__ U1,
                        const float* __restrict__ bias,
                        float* __restrict__ re, int L, int K, int cbase, int reW) {
  int b = blockIdx.x, k = threadIdx.x;
  const float* u0 = U0 + (long)b * L * K + k;
  const float* u1 = U1 + (long)b * L * K + k;
  float m = -1e30f;
  for (int l = 0; l < L - 1; ++l)
    m = fmaxf(m, fmaxf(u0[(long)l * K] + u1[(long)(l + 1) * K] + bias[k], 0.f));
  re[(long)b * reW + cbase + k] = fmaxf(m, 0.f);
}

// ---------------------------------------------------------------------------
// final dense [B,512] @ Wd^T[512,2] + bd, then log_softmax over 2 classes
// ---------------------------------------------------------------------------
__global__ void final_dense(const float* __restrict__ re, const float* __restrict__ Wd,
                            const float* __restrict__ bd, float* __restrict__ out, int D, int B) {
  int b = blockIdx.x * blockDim.x + threadIdx.x;
  if (b >= B) return;
  const float* r = re + (long)b * D;
  float z0 = bd[0], z1 = bd[1];
  for (int d = 0; d < D; ++d) { z0 += r[d] * Wd[d]; z1 += r[d] * Wd[D + d]; }
  float mx  = fmaxf(z0, z1);
  float lse = mx + __logf(__expf(z0 - mx) + __expf(z1 - mx));
  out[b * 2 + 0] = z0 - lse;
  out[b * 2 + 1] = z1 - lse;
}

// ---------------------------------------------------------------------------
extern "C" void kernel_launch(void* const* d_in, const int* in_sizes, int n_in,
                              void* d_out, int out_size, void* d_ws, size_t ws_size,
                              hipStream_t stream) {
  constexpr int N = 8192, H = 256, K = 128, B = 64, L = 128;
  constexpr int ML = B * L;        // 8192 gathered rows
  constexpr int D2 = 2 * H;        // 512

  const float* A_   = (const float*)d_in[0];
  const float* emb  = (const float*)d_in[1];
  const float* Wc   = (const float*)d_in[2];
  const float* bc   = (const float*)d_in[3];
  const float* W1   = (const float*)d_in[4];
  const float* b1   = (const float*)d_in[5];
  const float* W2   = (const float*)d_in[6];
  const float* b2   = (const float*)d_in[7];
  const float* Wd   = (const float*)d_in[8];
  const float* bd   = (const float*)d_in[9];
  const int* q_idx  = (const int*)d_in[10];
  const int* a_idx  = (const int*)d_in[11];
  float* out = (float*)d_out;

  // bump allocator over workspace
  char* ws = (char*)d_ws;
  size_t off = 0;
  auto alloc = [&](size_t bytes) -> void* {
    off = (off + 255) & ~(size_t)255;
    void* p = ws + off;
    off += bytes;
    return p;
  };

  bf16_t* Abf   = (bf16_t*)alloc((size_t)N * N * 2);
  bf16_t* Ebf   = (bf16_t*)alloc((size_t)N * H * 2);
  bf16_t* L1bf  = (bf16_t*)alloc((size_t)N * H * 2);
  bf16_t* Wcbf  = (bf16_t*)alloc((size_t)H * H * 2);
  bf16_t* W1bf  = (bf16_t*)alloc((size_t)K * D2 * 2);
  bf16_t* W2bf  = (bf16_t*)alloc((size_t)K * 2 * D2 * 2);
  float*  Qmf   = (float*)alloc((size_t)ML * H * 4);
  bf16_t* Qmbf  = (bf16_t*)alloc((size_t)ML * H * 2);
  float*  Amf   = (float*)alloc((size_t)ML * H * 4);
  bf16_t* Ambf  = (bf16_t*)alloc((size_t)ML * H * 2);
  float*  Sf    = (float*)alloc((size_t)B * L * L * 4);
  bf16_t* Pbf   = (bf16_t*)alloc((size_t)B * L * L * 2);
  bf16_t* P2bf  = (bf16_t*)alloc((size_t)B * L * L * 2);
  float*  Eqf   = (float*)alloc((size_t)ML * H * 4);
  float*  Eaf   = (float*)alloc((size_t)ML * H * 4);
  bf16_t* Tq    = (bf16_t*)alloc((size_t)ML * D2 * 2);
  bf16_t* Ta    = (bf16_t*)alloc((size_t)ML * D2 * 2);
  float*  Vq    = (float*)alloc((size_t)ML * K * 4);
  float*  U0q   = (float*)alloc((size_t)ML * K * 4);
  float*  U1q   = (float*)alloc((size_t)ML * K * 4);
  float*  Va    = (float*)alloc((size_t)ML * K * 4);
  float*  U0a   = (float*)alloc((size_t)ML * K * 4);
  float*  U1a   = (float*)alloc((size_t)ML * K * 4);
  float*  Re    = (float*)alloc((size_t)B * 4 * K * 4);
  (void)ws_size; (void)in_sizes; (void)n_in; (void)out_size;

  // --- 1. convert weights/activations to bf16 -------------------------------
  cvt_f32_bf16<<<32768, 256, 0, stream>>>((const float4*)A_,  (v4bf*)Abf,  (long)N * N / 4);
  cvt_f32_bf16<<<2048, 256, 0, stream>>>((const float4*)emb, (v4bf*)Ebf,  (long)N * H / 4);
  cvt_f32_bf16<<<64, 256, 0, stream>>>((const float4*)Wc,  (v4bf*)Wcbf, (long)H * H / 4);
  cvt_f32_bf16<<<64, 256, 0, stream>>>((const float4*)W1,  (v4bf*)W1bf, (long)K * D2 / 4);
  cvt_f32_bf16<<<128, 256, 0, stream>>>((const float4*)W2,  (v4bf*)W2bf, (long)K * 2 * D2 / 4);

  // --- 2. layer1 = A @ emb  (8192x8192x256) -> bf16 -------------------------
  wmma_gemm<false, false, false, false, false, true>
      <<<dim3(H / BN, N / BM, 1), 256, 0, stream>>>(
      Abf, 0, N, Ebf, 0, H, nullptr, L1bf, 0, H, nullptr, nullptr, N);

  // --- 3. gram: gather + relu(x @ Wc^T + bc) --------------------------------
  wmma_gemm<true, true, true, true, true, true>
      <<<dim3(H / BN, ML / BM, 1), 256, 0, stream>>>(
      L1bf, 0, H, Wcbf, 0, H, Qmf, Qmbf, 0, H, bc, q_idx, H);
  wmma_gemm<true, true, true, true, true, true>
      <<<dim3(H / BN, ML / BM, 1), 256, 0, stream>>>(
      L1bf, 0, H, Wcbf, 0, H, Amf, Ambf, 0, H, bc, a_idx, H);

  // --- 4. s[b] = Qm[b] @ Am[b]^T  (batched 128x128x256) ---------------------
  wmma_gemm<true, false, false, false, true, false>
      <<<dim3(L / BN, L / BM, B), 256, 0, stream>>>(
      Qmbf, (long)L * H, H, Ambf, (long)L * H, H,
      Sf, nullptr, (long)L * L, L, nullptr, nullptr, H);

  // --- 5. softmax row-wise (P) and column-wise (P2) -------------------------
  softmax128<<<dim3(L, B), L, 0, stream>>>(Sf, Pbf,  L, 1, (long)L * L, (long)L * L, L);
  softmax128<<<dim3(L, B), L, 0, stream>>>(Sf, P2bf, 1, L, (long)L * L, (long)L * L, L);

  // --- 6. eq = P @ Am ; ea = P2 @ Qm  (batched 128x256x128) -----------------
  wmma_gemm<false, false, false, false, true, false>
      <<<dim3(H / BN, L / BM, B), 256, 0, stream>>>(
      Pbf, (long)L * L, L, Ambf, (long)L * H, H,
      Eqf, nullptr, (long)L * H, H, nullptr, nullptr, L);
  wmma_gemm<false, false, false, false, true, false>
      <<<dim3(H / BN, L / BM, B), 256, 0, stream>>>(
      P2bf, (long)L * L, L, Qmbf, (long)L * H, H,
      Eaf, nullptr, (long)L * H, H, nullptr, nullptr, L);

  // --- 7. feature maps t = [(X-E)^2, X*E] -----------------------------------
  build_t<<<(ML * H + 255) / 256, 256, 0, stream>>>(Qmf, Eqf, Tq, (long)ML * H);
  build_t<<<(ML * H + 255) / 256, 256, 0, stream>>>(Amf, Eaf, Ta, (long)ML * H);

  // --- 8. conv projections (8192 x 128 x 512) -------------------------------
  wmma_gemm<true, false, false, false, true, false>
      <<<dim3(K / BN, ML / BM, 1), 256, 0, stream>>>(
      Tq, 0, D2, W1bf, 0, D2, Vq, nullptr, 0, K, nullptr, nullptr, D2);
  wmma_gemm<true, false, false, false, true, false>
      <<<dim3(K / BN, ML / BM, 1), 256, 0, stream>>>(
      Tq, 0, D2, W2bf, 0, 2 * D2, U0q, nullptr, 0, K, nullptr, nullptr, D2);
  wmma_gemm<true, false, false, false, true, false>
      <<<dim3(K / BN, ML / BM, 1), 256, 0, stream>>>(
      Tq, 0, D2, W2bf + D2, 0, 2 * D2, U1q, nullptr, 0, K, nullptr, nullptr, D2);
  wmma_gemm<true, false, false, false, true, false>
      <<<dim3(K / BN, ML / BM, 1), 256, 0, stream>>>(
      Ta, 0, D2, W1bf, 0, D2, Va, nullptr, 0, K, nullptr, nullptr, D2);
  wmma_gemm<true, false, false, false, true, false>
      <<<dim3(K / BN, ML / BM, 1), 256, 0, stream>>>(
      Ta, 0, D2, W2bf, 0, 2 * D2, U0a, nullptr, 0, K, nullptr, nullptr, D2);
  wmma_gemm<true, false, false, false, true, false>
      <<<dim3(K / BN, ML / BM, 1), 256, 0, stream>>>(
      Ta, 0, D2, W2bf + D2, 0, 2 * D2, U1a, nullptr, 0, K, nullptr, nullptr, D2);

  // --- 9. max-pools into re = [e1Q | e2Q | e1A | e2A] -----------------------
  pool_e1<<<B, K, 0, stream>>>(Vq,  b1, Re, L, K, 0 * K, 4 * K);
  pool_e2<<<B, K, 0, stream>>>(U0q, U1q, b2, Re, L, K, 1 * K, 4 * K);
  pool_e1<<<B, K, 0, stream>>>(Va,  b1, Re, L, K, 2 * K, 4 * K);
  pool_e2<<<B, K, 0, stream>>>(U0a, U1a, b2, Re, L, K, 3 * K, 4 * K);

  // --- 10. final dense + log_softmax ----------------------------------------
  final_dense<<<1, 64, 0, stream>>>(Re, Wd, bd, out, 4 * K, B);
}
